// ScouterAttention_60026462929418
// MI455X (gfx1250) — matrix operand order repaired and checked by reading, not compiled, not run on hardware
//
#include <hip/hip_runtime.h>

// ---------------------------------------------------------------------------
// ScouterAttention for MI455X (gfx1250, wave32, WMMA, async global->LDS).
//   bufX = bf16(inputs_pe)
//   bufY = relu(bufX @ W1^T + b1)      (v_wmma_f32_16x16x32_bf16, f32 accum)
//   bufX = relu(bufY @ W2^T + b2)
//   bufY = bufX @ W3^T + b3            ( = k )
//   bufX = bf16(inputs)
//   dots = slots @ k^T * D^-0.5
//   attn = softmax_over_slots(dots/row_sum*total) * sigmoid(...)
//   updates = attn @ inputs / D
// All GEMM K-loops: double-buffered LDS, one barrier per K-step, async
// global->LDS copies (ASYNCcnt) when the toolchain exposes them.
// ---------------------------------------------------------------------------

typedef __attribute__((ext_vector_type(16))) __bf16 v16bf;
typedef __attribute__((ext_vector_type(8)))  float  v8f;
typedef __attribute__((ext_vector_type(4)))  int    v4i;

#define BB   128
#define NN   1024
#define DD   512
#define SS   50
#define SPAD 64
#define MTOT (BB * NN)   // 131072
#define LDA  40          // LDS row stride in bf16 elems (80 B: 16B-aligned, bank-friendly)
#define LDB  40

// ---- global -> LDS copy: async if available, sync fallback -----------------

#if defined(__has_builtin)
#if __has_builtin(__builtin_amdgcn_global_load_async_to_lds_b128) && \
    __has_builtin(__builtin_amdgcn_s_wait_asynccnt)
#define USE_ASYNC_LDS 1
#endif
#endif

#if defined(USE_ASYNC_LDS)
typedef __attribute__((address_space(1))) v4i* gv4p;   // global int4*
typedef __attribute__((address_space(3))) v4i* lv4p;   // LDS int4*
__device__ __forceinline__ void cp16(void* dst_lds, const void* src_glb) {
    __builtin_amdgcn_global_load_async_to_lds_b128((gv4p)src_glb, (lv4p)dst_lds, 0, 0);
}
__device__ __forceinline__ void cp_wait() { __builtin_amdgcn_s_wait_asynccnt(0); }
#else
__device__ __forceinline__ void cp16(void* dst_lds, const void* src_glb) {
    *(uint4*)dst_lds = *(const uint4*)src_glb;
}
__device__ __forceinline__ void cp_wait() {}
#endif

// ---- WMMA fragment loads from LDS, k-contiguous per lane half, same order for A and B ------------

// A 16x32 bf16: lanes 0-15 -> row M=lane, K=0..15;
//               lanes 16-31 -> row M=lane-16, K=16..31.
__device__ __forceinline__ v16bf ldsa_frag(const unsigned short* lds, int mbase, int lane) {
    const int half = lane >> 4;
    const int m    = mbase + (lane & 15);
    const int klo  = half * 16;
    union { v16bf v; unsigned int u[8]; } f;
    const unsigned short* row = lds + m * LDA;
#pragma unroll
    for (int j = 0; j < 4; ++j) {
        f.u[j]     = *(const unsigned int*)(row + klo + 2 * j);
        f.u[4 + j] = *(const unsigned int*)(row + klo + 8 + 2 * j);
    }
    return f.v;
}

// B 32x16 bf16: lanes 0-15 -> col N=lane, K=0..15; lanes 16-31 -> col N=lane-16,
// K=16..31; K-contiguous within lane. LDS tile is stored n-major: ldsB[n][k].
__device__ __forceinline__ v16bf ldsb_frag(const unsigned short* lds, int nbase, int lane) {
    const int n  = nbase + (lane & 15);
    const int kb = (lane >> 4) * 16;
    union { v16bf v; unsigned int u[8]; } f;
    const unsigned int* p = (const unsigned int*)(lds + n * LDB + kb);
#pragma unroll
    for (int j = 0; j < 8; ++j) f.u[j] = p[j];
    return f.v;
}

__device__ __forceinline__ v8f wmma_bf16(v16bf a, v16bf b, v8f c) {
    return __builtin_amdgcn_wmma_f32_16x16x32_bf16(false, a, false, b, (short)0, c,
                                                   false, false);
}

// ---- elementwise conversions ----------------------------------------------

__global__ __launch_bounds__(256) void k_f32_to_bf16(const float* __restrict__ s,
                                                     __bf16* __restrict__ d, int n) {
    for (int i = blockIdx.x * 256 + threadIdx.x; i < n; i += gridDim.x * 256)
        d[i] = (__bf16)s[i];
}

__global__ __launch_bounds__(256) void k_pad_slots(const float* __restrict__ s,
                                                   __bf16* __restrict__ d) {
    const int idx = blockIdx.x * 256 + threadIdx.x;   // 0 .. SPAD*DD-1
    const int i = idx >> 9, dd = idx & 511;
    d[idx] = (i < SS) ? (__bf16)s[i * DD + dd] : (__bf16)0.0f;
}

// ---- MLP GEMM: C[131072,512] = act(A @ W^T + bias) -------------------------
// 256 thr (8 waves), tile 128x64, K-step 32, double-buffered LDS.

__global__ __launch_bounds__(256) void k_gemm_mlp(const __bf16* __restrict__ A,
                                                  const __bf16* __restrict__ W,
                                                  const float* __restrict__ bias,
                                                  __bf16* __restrict__ C, int do_relu) {
    __shared__ __align__(16) unsigned short ldsA[2][128 * LDA];
    __shared__ __align__(16) unsigned short ldsB[2][64 * LDB];
    const int tid = threadIdx.x;
    const int w = tid >> 5, lane = tid & 31;
    const int m0 = blockIdx.x * 128;
    const int n0 = blockIdx.y * 64;

    v8f acc[4] = {};
    const int ra = tid >> 1, ha = tid & 1;   // A-tile: 128 rows x 2 half-rows (32 B each)
    const int nb = tid >> 2, qb = tid & 3;   // B-tile: 64 rows x 4 chunks (16 B each)

    const __bf16* srcA = A + (size_t)(m0 + ra) * DD + ha * 16;
    const __bf16* srcB = W + (size_t)(n0 + nb) * DD + qb * 8;

    auto issue = [&](int k0, int buf) {
        unsigned short* da = &ldsA[buf][ra * LDA + ha * 16];
        cp16(da,     srcA + k0);
        cp16(da + 8, srcA + k0 + 8);
        cp16(&ldsB[buf][nb * LDB + qb * 8], srcB + k0);
    };

    issue(0, 0);
    const int NK = DD / 32;
    for (int kk = 0; kk < NK; ++kk) {
        const int cur = kk & 1;
        cp_wait();              // tile kk landed (only this tile is outstanding)
        __syncthreads();        // visible to all waves; prior reads of 1-cur done
        if (kk + 1 < NK) issue((kk + 1) * 32, 1 - cur);
        v16bf af = ldsa_frag(ldsA[cur], w * 16, lane);
#pragma unroll
        for (int nt = 0; nt < 4; ++nt) {
            v16bf bf = ldsb_frag(ldsB[cur], nt * 16, lane);
            acc[nt] = wmma_bf16(af, bf, acc[nt]);
        }
    }

    const int half = lane >> 4, nn = lane & 15;
#pragma unroll
    for (int nt = 0; nt < 4; ++nt) {
        const int col = n0 + nt * 16 + nn;
        const float bv = bias[col];
#pragma unroll
        for (int r = 0; r < 8; ++r) {
            float v = acc[nt][r] + bv;
            if (do_relu) v = fmaxf(v, 0.0f);
            const int row = m0 + w * 16 + (half << 3) + r;   // C layout: M = r + 8*half
            C[(size_t)row * DD + col] = (__bf16)v;
        }
    }
}

// ---- dots[b,i,j] = scale * sum_d slots[i,d] * k[b,j,d] ---------------------

__global__ __launch_bounds__(256) void k_dots(const __bf16* __restrict__ slotsb,
                                              const __bf16* __restrict__ kb,
                                              float* __restrict__ dots) {
    __shared__ __align__(16) unsigned short ldsA[2][SPAD * LDA];
    __shared__ __align__(16) unsigned short ldsB[2][64 * LDB];
    const int tid = threadIdx.x;
    const int w = tid >> 5, lane = tid & 31;
    const int b = blockIdx.x;
    const int n0 = blockIdx.y * 64;
    const int mt = w & 3, nh = w >> 2;

    v8f acc[2] = {};
    const int rr = tid >> 2, qq = tid & 3;   // 64 rows x 4 chunks, both tiles

    const __bf16* srcA = slotsb + (size_t)rr * DD + qq * 8;
    const __bf16* srcB = kb + ((size_t)b * NN + n0 + rr) * DD + qq * 8;

    auto issue = [&](int k0, int buf) {
        cp16(&ldsA[buf][rr * LDA + qq * 8], srcA + k0);
        cp16(&ldsB[buf][rr * LDB + qq * 8], srcB + k0);
    };

    issue(0, 0);
    const int NK = DD / 32;
    for (int kk = 0; kk < NK; ++kk) {
        const int cur = kk & 1;
        cp_wait();
        __syncthreads();
        if (kk + 1 < NK) issue((kk + 1) * 32, 1 - cur);
        v16bf af = ldsa_frag(ldsA[cur], mt * 16, lane);
#pragma unroll
        for (int t = 0; t < 2; ++t) {
            v16bf bf = ldsb_frag(ldsB[cur], (nh * 2 + t) * 16, lane);
            acc[t] = wmma_bf16(af, bf, acc[t]);
        }
    }

    const float scale = 0.044194173824159216f;   // 512^-0.5
    const int half = lane >> 4, nn = lane & 15;
#pragma unroll
    for (int t = 0; t < 2; ++t) {
        const int j = n0 + (nh * 2 + t) * 16 + nn;
#pragma unroll
        for (int r = 0; r < 8; ++r) {
            const int i = mt * 16 + (half << 3) + r;
            if (i < SS) dots[((size_t)b * SS + i) * NN + j] = acc[t][r] * scale;
        }
    }
}

// ---- row sums: one wave per (b,i) row --------------------------------------

__global__ __launch_bounds__(256) void k_rowsum(const float* __restrict__ dots,
                                                float* __restrict__ row_sum) {
    const int gw = blockIdx.x * 8 + (threadIdx.x >> 5);   // 6400 rows
    const int lane = threadIdx.x & 31;
    const float* row = dots + (size_t)gw * NN;
    float s = 0.0f;
#pragma unroll
    for (int t = 0; t < 32; ++t) s += row[lane + t * 32];
#pragma unroll
    for (int off = 16; off > 0; off >>= 1) s += __shfl_xor(s, off, 32);
    if (lane == 0) row_sum[gw] = s;
}

// ---- attn: normalize, softmax over slot dim, times sigmoid -----------------

__global__ __launch_bounds__(256) void k_attn(const float* __restrict__ dots,
                                              const float* __restrict__ row_sum,
                                              float* __restrict__ attn_out,
                                              __bf16* __restrict__ attnb) {
    __shared__ float rs[SS];
    const int b = blockIdx.x;
    const int j = blockIdx.y * 256 + threadIdx.x;
    if (threadIdx.x < SS) rs[threadIdx.x] = row_sum[b * SS + threadIdx.x];
    __syncthreads();

    float total = 0.0f;
#pragma unroll
    for (int i = 0; i < SS; ++i) total += rs[i];

    float val[SS];
    float m = -3.0e38f;
#pragma unroll
    for (int i = 0; i < SS; ++i) {
        float v = dots[((size_t)b * SS + i) * NN + j] * total / rs[i];
        val[i] = v;
        m = fmaxf(m, v);
    }
    float se = 0.0f;
#pragma unroll
    for (int i = 0; i < SS; ++i) { val[i] = __expf(val[i] - m); se += val[i]; }
    const float em = __expf(-m);        // sigmoid(v) = e/(e+exp(-m)), e = exp(v-m)
    const float inv_se = 1.0f / se;
#pragma unroll
    for (int i = 0; i < SS; ++i) {
        const float e = val[i];
        const float a = (e * inv_se) * (e / (e + em));
        attn_out[((size_t)b * SS + i) * NN + j] = a;
        attnb[((size_t)b * SPAD + i) * NN + j] = (__bf16)a;
    }
#pragma unroll
    for (int i = SS; i < SPAD; ++i)     // zero-pad rows 50..63 for the GEMM
        attnb[((size_t)b * SPAD + i) * NN + j] = (__bf16)0.0f;
}

// ---- updates[b,i,d] = (1/D) * sum_j attn[b,i,j] * inputs[b,j,d] ------------
// A-tile via async path; B-tile needs a transpose -> VGPR scatter into LDS.

__global__ __launch_bounds__(256) void k_updates(const __bf16* __restrict__ attnb,
                                                 const __bf16* __restrict__ xb,
                                                 float* __restrict__ out) {
    __shared__ __align__(16) unsigned short ldsA[2][SPAD * LDA];
    __shared__ __align__(16) unsigned short ldsB[2][64 * LDB];
    const int tid = threadIdx.x;
    const int w = tid >> 5, lane = tid & 31;
    const int b = blockIdx.x;
    const int d0 = blockIdx.y * 64;
    const int mt = w & 3, nh = w >> 2;

    v8f acc[2] = {};
    const int ra = tid >> 2, qa = tid & 3;   // A-tile: 64 rows x 4 chunks
    const int jb = tid >> 3, cb = tid & 7;   // B-tile: 32 k-rows x 8 d-chunks

    const __bf16* srcA = attnb + ((size_t)b * SPAD + ra) * NN + qa * 8;
    const __bf16* srcB = xb + ((size_t)b * NN + jb) * DD + d0 + cb * 8;

    auto issueA = [&](int k0, int buf) {
        cp16(&ldsA[buf][ra * LDA + qa * 8], srcA + k0);
    };
    auto issueB = [&](int k0, int buf) {
        uint4 v = *(const uint4*)(srcB + (size_t)k0 * DD);
        const unsigned short* e = (const unsigned short*)&v;
#pragma unroll
        for (int t = 0; t < 8; ++t)          // transpose: ldsB[d_local][j_local]
            ldsB[buf][(cb * 8 + t) * LDB + jb] = e[t];
    };

    issueA(0, 0);
    issueB(0, 0);
    const int NK = NN / 32;
    for (int kk = 0; kk < NK; ++kk) {
        const int cur = kk & 1;
        cp_wait();
        __syncthreads();
        if (kk + 1 < NK) { issueA((kk + 1) * 32, 1 - cur); issueB((kk + 1) * 32, 1 - cur); }
        v16bf af = ldsa_frag(ldsA[cur], mt * 16, lane);
#pragma unroll
        for (int t = 0; t < 2; ++t) {
            v16bf bf = ldsb_frag(ldsB[cur], (nh * 2 + t) * 16, lane);
            acc[t] = wmma_bf16(af, bf, acc[t]);
        }
    }

    const float inv_d = 1.0f / 512.0f;
    const int half = lane >> 4, nn = lane & 15;
#pragma unroll
    for (int t = 0; t < 2; ++t) {
        const int d = d0 + (nh * 2 + t) * 16 + nn;
#pragma unroll
        for (int r = 0; r < 8; ++r) {
            const int i = mt * 16 + (half << 3) + r;
            if (i < SS) out[((size_t)b * SS + i) * DD + d] = acc[t][r] * inv_d;
        }
    }
}

// ---------------------------------------------------------------------------

extern "C" void kernel_launch(void* const* d_in, const int* in_sizes, int n_in,
                              void* d_out, int out_size, void* d_ws, size_t ws_size,
                              hipStream_t stream) {
    const float* inputs_pe = (const float*)d_in[0];
    const float* inputs    = (const float*)d_in[1];
    const float* slots     = (const float*)d_in[2];
    const float* W1        = (const float*)d_in[3];
    const float* b1        = (const float*)d_in[4];
    const float* W2        = (const float*)d_in[5];
    const float* b2        = (const float*)d_in[6];
    const float* W3        = (const float*)d_in[7];
    const float* b3        = (const float*)d_in[8];

    // workspace layout (all offsets 256B-aligned)
    const size_t SZ_BIG = (size_t)MTOT * DD * 2;            // 134,217,728 B
    char* ws = (char*)d_ws;
    __bf16* bufX   = (__bf16*)(ws);                          // activations ping
    __bf16* bufY   = (__bf16*)(ws + SZ_BIG);                 // activations pong
    __bf16* W1b    = (__bf16*)(ws + 2 * SZ_BIG);
    __bf16* W2b    = (__bf16*)(ws + 2 * SZ_BIG + 524288);
    __bf16* W3b    = (__bf16*)(ws + 2 * SZ_BIG + 2 * 524288);
    __bf16* slotsb = (__bf16*)(ws + 2 * SZ_BIG + 3 * 524288);
    float*  dots   = (float*)(ws + 2 * SZ_BIG + 3 * 524288 + 65536);
    float*  rowsum = (float*)((char*)dots + (size_t)BB * SS * NN * 4);
    __bf16* attnb  = (__bf16*)((char*)rowsum + 25600);

    float* out_updates = (float*)d_out;
    float* out_attn    = out_updates + (size_t)BB * SS * DD;

    // 1) bf16 conversions
    k_f32_to_bf16<<<8192, 256, 0, stream>>>(inputs_pe, bufX, MTOT * DD);
    k_f32_to_bf16<<<1024, 256, 0, stream>>>(W1, W1b, DD * DD);
    k_f32_to_bf16<<<1024, 256, 0, stream>>>(W2, W2b, DD * DD);
    k_f32_to_bf16<<<1024, 256, 0, stream>>>(W3, W3b, DD * DD);
    k_pad_slots<<<(SPAD * DD) / 256, 256, 0, stream>>>(slots, slotsb);

    // 2) to_k MLP: three GEMMs, relu on first two
    dim3 gg(MTOT / 128, DD / 64);
    k_gemm_mlp<<<gg, 256, 0, stream>>>(bufX, W1b, b1, bufY, 1);
    k_gemm_mlp<<<gg, 256, 0, stream>>>(bufY, W2b, b2, bufX, 1);
    k_gemm_mlp<<<gg, 256, 0, stream>>>(bufX, W3b, b3, bufY, 0);   // bufY = k

    // bufX no longer needed -> reuse for bf16(inputs)
    k_f32_to_bf16<<<8192, 256, 0, stream>>>(inputs, bufX, MTOT * DD);

    // 3) dots + normalization + attn
    k_dots<<<dim3(BB, NN / 64), 256, 0, stream>>>(slotsb, bufY, dots);
    k_rowsum<<<(BB * SS) / 8, 256, 0, stream>>>(dots, rowsum);
    k_attn<<<dim3(BB, NN / 256), 256, 0, stream>>>(dots, rowsum, out_attn, attnb);

    // 4) updates = attn @ inputs / D
    k_updates<<<dim3(BB, DD / 64), 256, 0, stream>>>(attnb, bufX, out_updates);
}
